// MultiHeadAttention_69088843923801
// MI455X (gfx1250) — matrix operand order repaired and checked by reading, not compiled, not run on hardware
//
#include <hip/hip_runtime.h>
#include <hip/hip_bf16.h>

typedef __bf16 bf16_t;
typedef __attribute__((ext_vector_type(16))) __bf16 v16bf;
typedef __attribute__((ext_vector_type(8)))  __bf16 v8bf;
typedef __attribute__((ext_vector_type(8)))  float  v8f;

// ---------------- problem constants ----------------
#define BB   4
#define TT   2048
#define HH   256
#define NHD  8
#define NHH  2048      // NHD*HH
#define BT   8192      // BB*TT
#define FFD  1024      // 4*HH

// ---------------- WMMA helpers ----------------
static __device__ __forceinline__ v16bf cat8(v8bf lo, v8bf hi) {
  return __builtin_shufflevector(lo, hi, 0,1,2,3,4,5,6,7,8,9,10,11,12,13,14,15);
}

static __device__ __forceinline__ v8f zero_v8f() {
  v8f z;
#pragma unroll
  for (int i = 0; i < 8; ++i) z[i] = 0.0f;
  return z;
}

// A fragment (16x32 bf16, row-major, leading dim ld elements).
// Layout (ISA 7.12.2): lane m=l&15, half h=l>>4; regs0-3: K=h*8+0..7, regs4-7: K=16+h*8+0..7.
static __device__ __forceinline__ v16bf load_frag_a(const bf16_t* p0, int ld, int lane) {
  const int m = lane & 15, h = lane >> 4;
  const bf16_t* p = p0 + (size_t)m * ld + h * 8;
  v8bf lo = *(const v8bf*)p;
  v8bf hi = *(const v8bf*)(p + 16);
  return cat8(lo, hi);
}

// B fragment (32x16 KxN) from column-major storage Bt[n*ldk + k].
// lane n=l&15, half h=l>>4 -> regs0-3: K=h*8+0..7, regs4-7: K=16+h*8+0..7.
static __device__ __forceinline__ v16bf load_frag_b(const bf16_t* p0, int ldk, int lane) {
  const int n = lane & 15, h = lane >> 4;
  const bf16_t* p = p0 + (size_t)n * ldk + h * 8;
  v8bf lo = *(const v8bf*)p;
  v8bf hi = *(const v8bf*)(p + 16);
  return cat8(lo, hi);
}

static __device__ __forceinline__ v8f wmma_bf16(v16bf a, v16bf b, v8f c) {
  return __builtin_amdgcn_wmma_f32_16x16x32_bf16(false, a, false, b, (short)0, c, false, false);
}

// ---------------- async global->LDS copy primitives (CDNA5, ASYNCcnt) ----------------
// VDST = per-lane LDS byte offset (low 32 bits of a generic pointer to __shared__),
// VADDR = 64-bit global address. Tracked by ASYNCcnt.
static __device__ __forceinline__ void async_b128(unsigned loff, unsigned long long ga) {
  asm volatile("global_load_async_to_lds_b128 %0, %1, off" :: "v"(loff), "v"(ga) : "memory");
}
static __device__ __forceinline__ unsigned lds_off(const void* p) {
  return (unsigned)(unsigned long long)p;   // LDS aperture base has zero low 32 bits
}
static __device__ __forceinline__ void wait_async0() {
  asm volatile("s_wait_asynccnt 0" ::: "memory");
}

// ---------------- conversion / transpose kernels ----------------
__global__ void cvt_bf16_kernel(const float* __restrict__ in, bf16_t* __restrict__ out, int n) {
  int i = blockIdx.x * 256 + threadIdx.x;
  if (i < n) out[i] = (bf16_t)in[i];
}

// W (K x N row-major, f32) -> Wt (N x K row-major, bf16) == column-major B
__global__ void cvt_transpose_kernel(const float* __restrict__ W, bf16_t* __restrict__ Wt,
                                     int K, int N) {
  int i = blockIdx.x * 256 + threadIdx.x;
  if (i < K * N) {
    int n = i / K, k = i - n * K;
    Wt[i] = (bf16_t)W[(size_t)k * N + n];
  }
}

// Vb per head (2048 x 256 row-major) -> Vt per head (256 x 2048 row-major)
__global__ void transpose_v_kernel(const bf16_t* __restrict__ Vb, bf16_t* __restrict__ Vt) {
  size_t idx = (size_t)blockIdx.x * 256 + threadIdx.x;   // total 16777216
  int t  = (int)(idx & 2047);
  int h  = (int)((idx >> 11) & 255);
  int bh = (int)(idx >> 19);
  Vt[idx] = Vb[(size_t)bh * 524288 + (size_t)t * 256 + h];
}

// ---------------- generic WMMA GEMM with async-LDS-staged B tiles ----------------
// C(M,N) = A(MxK bf16 rm) * B(KxN), B stored column-major in Bt(NxK bf16 rm).
// 8 waves/block; wave -> 16 rows; block -> 128 rows x 64 cols.
// B tile (64 cols x 32 k = 4KB) staged to LDS once per block, double-buffered.
template <bool BIAS, bool RELU, bool OUT16>
__global__ void __launch_bounds__(256)
gemm_wmma_kernel(const bf16_t* __restrict__ A, const bf16_t* __restrict__ Bt,
                 const float* __restrict__ bias, void* __restrict__ Cout,
                 int M, int N, int K) {
  const int tid  = threadIdx.x;
  const int lane = tid & 31;
  const int wave = tid >> 5;
  const int row0 = blockIdx.y * 128 + wave * 16;
  const int col0 = blockIdx.x * 64;
  (void)M;

  __shared__ __align__(16) bf16_t bbuf[2][64 * 32];   // 2 x 4KB

  // one 16B async transfer per thread per chunk: tile row n = tid>>2, quarter q = tid&3
  const int bn = tid >> 2, bq = tid & 3;
  const unsigned bl0 = lds_off(&bbuf[0][0]) + (unsigned)tid * 16;
  const unsigned bl1 = lds_off(&bbuf[1][0]) + (unsigned)tid * 16;
  const unsigned long long bga =
      (unsigned long long)(const char*)(Bt + (size_t)(col0 + bn) * K) + (unsigned)bq * 16;

  v8f acc[4];
#pragma unroll
  for (int t = 0; t < 4; ++t) acc[t] = zero_v8f();

  // prologue: prefetch kb=0 into buffer 0
  async_b128(bl0, bga);

  const bf16_t* arow = A + (size_t)row0 * K;
  for (int kb = 0; kb < K; kb += 32) {
    const int cur = (kb >> 5) & 1;
    wait_async0();
    __syncthreads();
    if (kb + 32 < K)
      async_b128(cur ? bl0 : bl1, bga + (unsigned)(kb + 32) * 2);

    v16bf a = load_frag_a(arow + kb, K, lane);
#pragma unroll
    for (int t = 0; t < 4; ++t) {
      v16bf b = load_frag_b(&bbuf[cur][(t * 16) * 32], 32, lane);
      acc[t] = wmma_bf16(a, b, acc[t]);
    }
  }

  const int h = lane >> 4, cl = lane & 15;
#pragma unroll
  for (int t = 0; t < 4; ++t) {
    int col = col0 + t * 16 + cl;
    float bv = BIAS ? bias[col] : 0.0f;
#pragma unroll
    for (int r = 0; r < 8; ++r) {
      float v = acc[t][r] + bv;
      if (RELU) v = fmaxf(v, 0.0f);
      size_t o = (size_t)(row0 + r + 8 * h) * N + col;
      if (OUT16) ((bf16_t*)Cout)[o] = (bf16_t)v;
      else       ((float*)Cout)[o]  = v;
    }
  }
}

// ---------------- flash attention with async-LDS-staged K/V tiles ----------------
// grid = (T/64, NH, B), block = 128 (4 waves). Wave handles 16 query rows.
// Per key chunk (32 keys): K tile 32x256 bf16 (16KB, contiguous in global),
// V^T tile 256x32 bf16 (16KB, 64B rows strided by 4KB). Double-buffered in LDS.
__global__ void __launch_bounds__(128)
attn_wmma_kernel(const bf16_t* __restrict__ Q, const bf16_t* __restrict__ Km,
                 const bf16_t* __restrict__ Vt, bf16_t* __restrict__ O) {
  const int tid  = threadIdx.x;
  const int lane = tid & 31;
  const int wave = tid >> 5;
  const int b = blockIdx.z, n = blockIdx.y;
  const int qrow = blockIdx.x * 64 + wave * 16;          // query row within head
  const size_t head = ((size_t)b * NHD + n) * ((size_t)TT * HH);
  const bf16_t* Qh = Q  + head;     // (2048 x 256) rm
  const bf16_t* Kh = Km + head;     // (2048 x 256) rm == (K^T) column-major
  const bf16_t* Vh = Vt + head;     // (256 x 2048) rm == V column-major
  const int cl = lane & 15, hh = lane >> 4;

  __shared__ __align__(16) bf16_t kbuf[2][32 * 256];    // 2 x 16KB
  __shared__ __align__(16) bf16_t vbuf[2][256 * 32];    // 2 x 16KB
  __shared__ __align__(16) bf16_t pbuf[4][16][40];      // per-wave P transpose tile

  // async copy helpers for this thread's share (8 x 16B each for K and V)
  const unsigned kl[2] = { lds_off(&kbuf[0][0]), lds_off(&kbuf[1][0]) };
  const unsigned vl[2] = { lds_off(&vbuf[0][0]), lds_off(&vbuf[1][0]) };
  const unsigned long long kg = (unsigned long long)(const char*)Kh;
  const unsigned long long vg = (unsigned long long)(const char*)Vh;

  // K tile: contiguous 16KB at Kh + jb*256 ; transfer idx = tid + i*128
  // V tile: row = idx>>2 (dim), quarter = idx&3 ; global row stride = TT elems
#define ISSUE_TILES(buf, jb_)                                                          \
  {                                                                                    \
    _Pragma("unroll")                                                                  \
    for (int i = 0; i < 8; ++i) {                                                      \
      int idx = tid + i * 128;                                                         \
      async_b128(kl[buf] + (unsigned)idx * 16,                                         \
                 kg + ((unsigned long long)(jb_) * HH + (unsigned long long)idx * 8) * 2); \
    }                                                                                  \
    _Pragma("unroll")                                                                  \
    for (int i = 0; i < 8; ++i) {                                                      \
      int idx = tid + i * 128;                                                         \
      int row = idx >> 2, q = idx & 3;                                                 \
      async_b128(vl[buf] + (unsigned)idx * 16,                                         \
                 vg + ((unsigned long long)row * TT + (jb_) + (unsigned long long)q * 8) * 2); \
    }                                                                                  \
  }

  // preload Q fragments for this wave's 16 rows (K = 256 in 8 chunks)
  v16bf qf[8];
#pragma unroll
  for (int kk = 0; kk < 8; ++kk)
    qf[kk] = load_frag_a(Qh + (size_t)qrow * HH + kk * 32, HH, lane);

  v8f oacc[16];
#pragma unroll
  for (int f = 0; f < 16; ++f) oacc[f] = zero_v8f();
  float mx[8], sm[8];
#pragma unroll
  for (int r = 0; r < 8; ++r) { mx[r] = -3.0e38f; sm[r] = 0.0f; }

  const float scale = 0.0625f;  // 1/sqrt(256)

  // prologue: prefetch chunk 0 into buffer 0
  ISSUE_TILES(0, 0)

  for (int jb = 0; jb < TT; jb += 32) {
    const int cur = (jb >> 5) & 1;
    wait_async0();          // own wave's async copies into buf[cur] complete
    __syncthreads();        // all waves' shares done; other buffer free for reuse
    if (jb + 32 < TT)
      ISSUE_TILES(cur ^ 1, jb + 32)

    // ---- S = Q * K^T for 32 keys (two 16x16 fragments), K tile from LDS ----
    v8f s0 = zero_v8f(), s1 = zero_v8f();
#pragma unroll
    for (int kk = 0; kk < 8; ++kk) {
      v16bf b0 = load_frag_b(&kbuf[cur][kk * 32], HH, lane);
      v16bf b1 = load_frag_b(&kbuf[cur][16 * HH + kk * 32], HH, lane);
      s0 = wmma_bf16(qf[kk], b0, s0);
      s1 = wmma_bf16(qf[kk], b1, s1);
    }

    // ---- online softmax update (row = r + 8*hh, reduce across 16-lane half) ----
    float p0[8], p1[8];
#pragma unroll
    for (int r = 0; r < 8; ++r) {
      float a0 = s0[r] * scale, a1 = s1[r] * scale;
      float cm = fmaxf(a0, a1);
      for (int msk = 1; msk < 16; msk <<= 1) cm = fmaxf(cm, __shfl_xor(cm, msk, 32));
      float mnew  = fmaxf(mx[r], cm);
      float alpha = __expf(mx[r] - mnew);
      float e0 = __expf(a0 - mnew), e1 = __expf(a1 - mnew);
      float rs = e0 + e1;
      for (int msk = 1; msk < 16; msk <<= 1) rs += __shfl_xor(rs, msk, 32);
      sm[r] = sm[r] * alpha + rs;
      mx[r] = mnew;
      p0[r] = e0; p1[r] = e1;
#pragma unroll
      for (int f = 0; f < 16; ++f) oacc[f][r] *= alpha;
    }

    // ---- transpose P from C-layout to A-layout via per-wave LDS tile ----
#pragma unroll
    for (int r = 0; r < 8; ++r) {
      pbuf[wave][r + 8 * hh][cl]      = (bf16_t)p0[r];
      pbuf[wave][r + 8 * hh][16 + cl] = (bf16_t)p1[r];
    }
    asm volatile("s_wait_dscnt 0" ::: "memory");   // wave-local LDS RAW fence
    v8bf plo = *(const v8bf*)&pbuf[wave][cl][hh * 8];
    v8bf phi = *(const v8bf*)&pbuf[wave][cl][16 + hh * 8];
    v16bf pa = cat8(plo, phi);
    asm volatile("" ::: "memory");

    // ---- O += P * V, V^T tile from LDS (ldk = 32) ----
#pragma unroll
    for (int f = 0; f < 16; ++f) {
      v16bf vb = load_frag_b(&vbuf[cur][(f * 16) * 32], 32, lane);
      oacc[f] = wmma_bf16(pa, vb, oacc[f]);
    }
  }
#undef ISSUE_TILES

  // ---- normalize and store (same flat layout as GEMM output) ----
#pragma unroll
  for (int r = 0; r < 8; ++r) {
    float inv = 1.0f / sm[r];
    int row = qrow + r + 8 * hh;
#pragma unroll
    for (int f = 0; f < 16; ++f)
      O[head + (size_t)row * HH + f * 16 + cl] = (bf16_t)(oacc[f][r] * inv);
  }
}

// ---------------- residual + layernorm ----------------
__global__ void __launch_bounds__(256)
ln_kernel(const float* __restrict__ a, const float* __restrict__ bres,
          const float* __restrict__ g, const float* __restrict__ beta,
          float* __restrict__ out, bf16_t* __restrict__ out16) {
  int row = blockIdx.x;
  int i = threadIdx.x;
  float v = a[(size_t)row * HH + i] + bres[(size_t)row * HH + i];

  float s = v, s2 = v * v;
  for (int m = 1; m < 32; m <<= 1) { s += __shfl_xor(s, m, 32); s2 += __shfl_xor(s2, m, 32); }
  __shared__ float red[2][8];
  int w = i >> 5, l = i & 31;
  if (l == 0) { red[0][w] = s; red[1][w] = s2; }
  __syncthreads();
  float ts = 0.0f, ts2 = 0.0f;
#pragma unroll
  for (int k = 0; k < 8; ++k) { ts += red[0][k]; ts2 += red[1][k]; }
  float mu  = ts * (1.0f / HH);
  float var = ts2 * (1.0f / HH) - mu * mu;
  float rs  = rsqrtf(var + 1e-5f);
  float y = (v - mu) * rs * g[i] + beta[i];
  out[(size_t)row * HH + i] = y;
  if (out16 != nullptr) out16[(size_t)row * HH + i] = (bf16_t)y;
}

// ---------------- workspace layout (bytes) ----------------
#define SZ_XB   ((size_t)BT * HH * 2)          // 4 MB
#define SZ_WBIG ((size_t)NHH * HH * 2)         // 1 MB
#define SZ_WFF  ((size_t)FFD * HH * 2)         // 0.5 MB
#define SZ_QKV  ((size_t)BT * NHH * 2)         // 32 MB
#define SZ_TMP  ((size_t)BT * HH * 4)          // 8 MB

#define O_XB   ((size_t)0)
#define O_WQ   (O_XB + SZ_XB)
#define O_WK   (O_WQ + SZ_WBIG)
#define O_WV   (O_WK + SZ_WBIG)
#define O_WO   (O_WV + SZ_WBIG)
#define O_W1   (O_WO + SZ_WBIG)
#define O_W2   (O_W1 + SZ_WFF)
#define O_QB   (O_W2 + SZ_WFF)
#define O_KB   (O_QB + SZ_QKV)
#define O_VB   (O_KB + SZ_QKV)
#define O_VT   (O_VB + SZ_QKV)
#define O_OB   (O_VT + SZ_QKV)
#define O_TMP  (O_OB + SZ_QKV)
#define O_H1   (O_TMP + SZ_TMP)
#define O_H1B  (O_H1 + SZ_TMP)
#define O_FF   (O_H1B + SZ_XB)

extern "C" void kernel_launch(void* const* d_in, const int* in_sizes, int n_in,
                              void* d_out, int out_size, void* d_ws, size_t ws_size,
                              hipStream_t stream) {
  (void)in_sizes; (void)n_in; (void)out_size; (void)ws_size;
  const float* x   = (const float*)d_in[0];
  const float* Wq  = (const float*)d_in[1];
  const float* Wk  = (const float*)d_in[2];
  const float* Wv  = (const float*)d_in[3];
  const float* Wo  = (const float*)d_in[4];
  const float* g1  = (const float*)d_in[5];
  const float* b1  = (const float*)d_in[6];
  const float* W1  = (const float*)d_in[7];
  const float* bb1 = (const float*)d_in[8];
  const float* W2  = (const float*)d_in[9];
  const float* bb2 = (const float*)d_in[10];
  const float* g2  = (const float*)d_in[11];
  const float* b2  = (const float*)d_in[12];
  float* out = (float*)d_out;

  char* ws = (char*)d_ws;
  bf16_t* xb  = (bf16_t*)(ws + O_XB);
  bf16_t* wqt = (bf16_t*)(ws + O_WQ);
  bf16_t* wkt = (bf16_t*)(ws + O_WK);
  bf16_t* wvt = (bf16_t*)(ws + O_WV);
  bf16_t* wot = (bf16_t*)(ws + O_WO);
  bf16_t* w1t = (bf16_t*)(ws + O_W1);
  bf16_t* w2t = (bf16_t*)(ws + O_W2);
  bf16_t* Qb  = (bf16_t*)(ws + O_QB);
  bf16_t* Kb  = (bf16_t*)(ws + O_KB);
  bf16_t* Vb  = (bf16_t*)(ws + O_VB);
  bf16_t* Vt  = (bf16_t*)(ws + O_VT);
  bf16_t* Ob  = (bf16_t*)(ws + O_OB);
  float*  tmp = (float*)(ws + O_TMP);
  float*  h1  = (float*)(ws + O_H1);
  bf16_t* h1b = (bf16_t*)(ws + O_H1B);
  bf16_t* ff  = (bf16_t*)(ws + O_FF);

  const dim3 blk256(256), blk128(128);

  // 1) input / weight precision conversion (+ weight transpose to column-major)
  cvt_bf16_kernel<<<dim3((BT * HH + 255) / 256), blk256, 0, stream>>>(x, xb, BT * HH);
  cvt_transpose_kernel<<<dim3((HH * NHH + 255) / 256), blk256, 0, stream>>>(Wq, wqt, HH, NHH);
  cvt_transpose_kernel<<<dim3((HH * NHH + 255) / 256), blk256, 0, stream>>>(Wk, wkt, HH, NHH);
  cvt_transpose_kernel<<<dim3((HH * NHH + 255) / 256), blk256, 0, stream>>>(Wv, wvt, HH, NHH);
  cvt_transpose_kernel<<<dim3((NHH * HH + 255) / 256), blk256, 0, stream>>>(Wo, wot, NHH, HH);
  cvt_transpose_kernel<<<dim3((HH * FFD + 255) / 256), blk256, 0, stream>>>(W1, w1t, HH, FFD);
  cvt_transpose_kernel<<<dim3((FFD * HH + 255) / 256), blk256, 0, stream>>>(W2, w2t, FFD, HH);

  // 2) QKV projections (WMMA, async-staged B tiles)
  gemm_wmma_kernel<false, false, true><<<dim3(NHH / 64, BT / 128), blk256, 0, stream>>>(
      xb, wqt, nullptr, Qb, BT, NHH, HH);
  gemm_wmma_kernel<false, false, true><<<dim3(NHH / 64, BT / 128), blk256, 0, stream>>>(
      xb, wkt, nullptr, Kb, BT, NHH, HH);
  gemm_wmma_kernel<false, false, true><<<dim3(NHH / 64, BT / 128), blk256, 0, stream>>>(
      xb, wvt, nullptr, Vb, BT, NHH, HH);

  // 3) V -> V^T per head (for contiguous PV B-fragments)
  transpose_v_kernel<<<dim3((BT * NHH) / 256), blk256, 0, stream>>>(Vb, Vt);

  // 4) flash attention (async double-buffered K/V tiles + WMMA + online softmax)
  attn_wmma_kernel<<<dim3(TT / 64, NHD, BB), blk128, 0, stream>>>(Qb, Kb, Vt, Ob);

  // 5) output projection (f32 out) + residual + LN1 (emits bf16 copy)
  gemm_wmma_kernel<false, false, false><<<dim3(HH / 64, BT / 128), blk256, 0, stream>>>(
      Ob, wot, nullptr, tmp, BT, HH, NHH);
  ln_kernel<<<dim3(BT), blk256, 0, stream>>>(x, tmp, g1, b1, h1, h1b);

  // 6) FFN: relu(h1 @ W1 + bb1) @ W2 + bb2, then residual + LN2 -> out
  gemm_wmma_kernel<true, true, true><<<dim3(FFD / 64, BT / 128), blk256, 0, stream>>>(
      h1b, w1t, bb1, ff, BT, FFD, HH);
  gemm_wmma_kernel<true, false, false><<<dim3(HH / 64, BT / 128), blk256, 0, stream>>>(
      ff, w2t, bb2, tmp, BT, HH, FFD);
  ln_kernel<<<dim3(BT), blk256, 0, stream>>>(h1, tmp, g2, b2, out, nullptr);
}
